// REAN_73942156968311
// MI455X (gfx1250) — matrix operand order repaired and checked by reading, not compiled, not run on hardware
//
#include <hip/hip_runtime.h>
#include <cstdint>
#include <cstddef>

// ---------------- types for CDNA5 WMMA ----------------
typedef __bf16 bf16_t;
typedef __bf16 v16bf __attribute__((ext_vector_type(16)));
typedef float  v8f   __attribute__((ext_vector_type(8)));

// ---------------- problem constants ----------------
#define NB     8
#define TOK    1024
#define DIM    768
#define DCATD  1536
#define NHEAD  3
#define DKH    512
#define CONVN  766

// ---------------- GEMM tiling ----------------
#define BM  128
#define BN  64
#define BKK 64

enum { EPI_NONE = 0, EPI_BIAS = 1, EPI_RELU = 3, EPI_SCALE = 4, EPI_GCN = 5, EPI_ROWBIAS_RELU = 6 };

__device__ __forceinline__ float waveMax(float v) {
#pragma unroll
  for (int off = 16; off; off >>= 1) v = fmaxf(v, __shfl_xor(v, off, 32));
  return v;
}
__device__ __forceinline__ float waveSum(float v) {
#pragma unroll
  for (int off = 16; off; off >>= 1) v += __shfl_xor(v, off, 32);
  return v;
}
__device__ __forceinline__ int waveSumI(int v) {
#pragma unroll
  for (int off = 16; off; off >>= 1) v += __shfl_xor(v, off, 32);
  return v;
}

// =====================================================================
// Generic bf16-WMMA GEMM: C[M,N] = epi(A[M,K] * B[K,N])
// BMODE 0: B row-major [K,N]
// BMODE 1: B transposed, global is [N,K] (ldb = row stride of B^T)
// BMODE 2: implicit-im2col conv B: B'[c,n] = X[(c/3)*ldb + n + (c%3)]
// Batched over blockIdx.z with (z/heads, z%heads) strides for A/B.
// =====================================================================
template <int BMODE>
__global__ __launch_bounds__(256) void gemm_wmma_bf16(
    const float* __restrict__ A, const float* __restrict__ Bm, float* __restrict__ C,
    int M, int N, int K, int lda, int ldb, int ldc,
    long long sAb, long long sAh, long long sBb, long long sBh, long long sCz,
    int heads, int epi,
    const float* __restrict__ bias, const float* __restrict__ denom, long long sDen,
    const float* __restrict__ add, long long sAdd, float scale)
{
  __shared__ bf16_t As[BM][BKK];   // 16 KB, A tile row-major [m][k]
  __shared__ bf16_t Bs[BN][BKK];   //  8 KB, B tile transposed [n][k]

  const int z  = blockIdx.z;
  const int bz = z / heads;
  const int hz = z - bz * heads;
  const float* Az = A + (long long)bz * sAb + (long long)hz * sAh;
  const float* Bz = Bm + (long long)bz * sBb + (long long)hz * sBh;
  float* Cz = C + (long long)z * sCz;

  const int mBase = blockIdx.y * BM;
  const int nBase = blockIdx.x * BN;

  const int tid   = threadIdx.x;
  const int lane  = tid & 31;
  const int wave  = tid >> 5;
  const int wm    = wave & 3;   // 4 waves along M (32 rows each)
  const int wn    = wave >> 2;  // 2 waves along N (32 cols each)
  const int lrow  = lane & 15;
  const int lhalf = lane >> 4;

  v8f acc[2][2];
#pragma unroll
  for (int i = 0; i < 2; ++i)
#pragma unroll
    for (int j = 0; j < 2; ++j)
#pragma unroll
      for (int t = 0; t < 8; ++t) acc[i][j][t] = 0.f;

  for (int k0 = 0; k0 < K; k0 += BKK) {
    // ---- A tile load: 128x64 f32 -> bf16, 32 elements per thread ----
    {
      const int arow = tid >> 1;
      const int kb   = (tid & 1) * 32;
      bf16_t* dst = &As[arow][kb];
      if (mBase + arow < M) {
        const float* src = Az + (size_t)(mBase + arow) * lda + (size_t)(k0 + kb);
        if (k0 + kb + 32 <= K) {
          __builtin_prefetch(src + BKK, 0, 1);   // gfx1250 global_prefetch_b8
#pragma unroll
          for (int t = 0; t < 16; ++t) {
            float2 v = ((const float2*)src)[t];  // all ldas even -> 8B aligned
            dst[2 * t]     = (bf16_t)v.x;
            dst[2 * t + 1] = (bf16_t)v.y;
          }
        } else {
          for (int t = 0; t < 32; ++t) {
            int kg = k0 + kb + t;
            dst[t] = (kg < K) ? (bf16_t)src[t] : (bf16_t)0.f;
          }
        }
      } else {
        for (int t = 0; t < 32; ++t) dst[t] = (bf16_t)0.f;
      }
    }
    // ---- B tile load into transposed LDS layout Bs[n][k] ----
    if (BMODE == 0) {
      const int krow  = tid >> 2;          // 0..63 (k within tile)
      const int nb    = (tid & 3) * 16;    // 16 n per thread
      const int kg    = k0 + krow;
      if (kg < K && nBase + nb + 16 <= N) {
        const float* src = Bz + (size_t)kg * ldb + (size_t)(nBase + nb);
#pragma unroll
        for (int t = 0; t < 8; ++t) {
          float2 v = ((const float2*)src)[t];
          Bs[nb + 2 * t][krow]     = (bf16_t)v.x;
          Bs[nb + 2 * t + 1][krow] = (bf16_t)v.y;
        }
      } else {
        for (int t = 0; t < 16; ++t) {
          int nc = nBase + nb + t;
          Bs[nb + t][krow] = (kg < K && nc < N) ? (bf16_t)Bz[(size_t)kg * ldb + nc]
                                                : (bf16_t)0.f;
        }
      }
    } else if (BMODE == 1) {
      const int nrow = tid >> 2;           // 0..63 (n within tile)
      const int kbp  = (tid & 3) * 16;     // 16 k per thread
      const int ng   = nBase + nrow;
      if (ng < N && k0 + kbp + 16 <= K) {
        const float* src = Bz + (size_t)ng * ldb + (size_t)(k0 + kbp);
#pragma unroll
        for (int t = 0; t < 8; ++t) {
          float2 v = ((const float2*)src)[t];
          Bs[nrow][kbp + 2 * t]     = (bf16_t)v.x;
          Bs[nrow][kbp + 2 * t + 1] = (bf16_t)v.y;
        }
      } else {
        for (int t = 0; t < 16; ++t) {
          int kg = k0 + kbp + t;
          Bs[nrow][kbp + t] = (ng < N && kg < K) ? (bf16_t)Bz[(size_t)ng * ldb + kg]
                                                 : (bf16_t)0.f;
        }
      }
    } else { // BMODE == 2: conv implicit im2col
      const int krow = tid >> 2;           // 0..63
      const int nb   = (tid & 3) * 16;
      const int c    = k0 + krow;          // K = 3072, multiple of 64
      const int ic   = c / 3;
      const int kk   = c - 3 * ic;
      const float* src = Bz + (size_t)ic * ldb + kk;
      for (int t = 0; t < 16; ++t) {
        int nc = nBase + nb + t;
        Bs[nb + t][krow] = (nc < N) ? (bf16_t)src[nc] : (bf16_t)0.f;
      }
    }
    __syncthreads();

    // ---- compute: 2 k-steps of 32, 2x2 WMMA tiles per wave ----
#pragma unroll
    for (int kk = 0; kk < BKK; kk += 32) {
      v16bf af[2], bfm[2];
#pragma unroll
      for (int i = 0; i < 2; ++i) {
        // A 16x32 bf16 fragment (ISA 7.12.2): lane half selects K chunk
        const bf16_t* ap = &As[wm * 32 + i * 16 + lrow][kk + lhalf * 8];
#pragma unroll
        for (int j = 0; j < 8; ++j) { af[i][j] = ap[j]; af[i][8 + j] = ap[16 + j]; }
      }
#pragma unroll
      for (int i = 0; i < 2; ++i) {
        // B 32x16 fragment: lane holds col N=lrow, lane half selects K chunk (same K order as A)
        const bf16_t* bp = &Bs[wn * 32 + i * 16 + lrow][kk + lhalf * 8];
#pragma unroll
        for (int j = 0; j < 8; ++j) { bfm[i][j] = bp[j]; bfm[i][8 + j] = bp[16 + j]; }
      }
#pragma unroll
      for (int i = 0; i < 2; ++i)
#pragma unroll
        for (int j = 0; j < 2; ++j)
          acc[i][j] = __builtin_amdgcn_wmma_f32_16x16x32_bf16(
              false, af[i], false, bfm[j], (short)0, acc[i][j], false, false);
    }
    __syncthreads();
  }

  // ---- epilogue ----
  const float* denZ = (denom != nullptr) ? denom + (size_t)bz * sDen : nullptr;
  const float* addZ = (add   != nullptr) ? add   + (size_t)z  * sAdd : nullptr;
#pragma unroll
  for (int i = 0; i < 2; ++i) {
#pragma unroll
    for (int j = 0; j < 2; ++j) {
      const int baseR = mBase + wm * 32 + i * 16 + lhalf * 8;
      const int baseC = nBase + wn * 32 + j * 16 + lrow;
      if (baseC < N) {
#pragma unroll
        for (int r = 0; r < 8; ++r) {
          const int row = baseR + r;
          if (row < M) {
            float v = acc[i][j][r];
            if (epi == EPI_BIAS)             v += bias[baseC];
            else if (epi == EPI_RELU)        v = fmaxf(v, 0.f);
            else if (epi == EPI_SCALE)       v *= scale;
            else if (epi == EPI_GCN)         v = fmaxf((v + bias[baseC]) / denZ[row], 0.f)
                                                 + addZ[(size_t)row * ldc + baseC];
            else if (epi == EPI_ROWBIAS_RELU) v = fmaxf(v + bias[row], 0.f);
            Cz[(size_t)row * ldc + baseC] = v;
          }
        }
      }
    }
  }
}

// =====================================================================
// small helper kernels
// =====================================================================
__global__ __launch_bounds__(256) void rowsum_kernel(const float* __restrict__ src,
                                                     float* __restrict__ dst, int width) {
  const int row  = blockIdx.x * 8 + (threadIdx.x >> 5);
  const int lane = threadIdx.x & 31;
  const float* p = src + (size_t)row * width;
  float s = 0.f;
  for (int c = lane; c < width; c += 32) s += p[c];
  s = waveSum(s);
  if (lane == 0) dst[row] = s + 1.0f;
}

__global__ __launch_bounds__(256) void concat_inputs_kernel(const float* __restrict__ inp,
                                                            float* __restrict__ x) {
  const size_t i4  = (size_t)blockIdx.x * 256 + threadIdx.x;
  const size_t row = i4 / (DIM / 4);
  const size_t c   = (i4 % (DIM / 4)) * 4;
  float4 v = *(const float4*)(inp + row * DIM + c);
  *(float4*)(x + row * (size_t)DCATD + c) = v;
}

__global__ void init_small_kernel(float* headsum, unsigned* m1, unsigned* m2,
                                  unsigned* cnt, int* maxidx) {
  const int t = threadIdx.x;
  if (t < NB * NHEAD) headsum[t] = 0.f;
  if (t < NB) { m1[t] = 0u; m2[t] = 0u; cnt[t] = 0u; maxidx[t] = 0; }
}

// one block per (b,h,q) row of scores; in-place masked softmax * keep
__global__ __launch_bounds__(256) void softmax_mask_kernel(float* __restrict__ scores,
                                                           const unsigned char* __restrict__ mask,
                                                           float* __restrict__ headsum) {
  const int row = blockIdx.x;              // ((b*NHEAD+h)*TOK + q)
  const int bh  = row >> 10;
  float* s = scores + (size_t)row * TOK;
  const unsigned char* m = mask + (size_t)row * TOK;
  const int tid = threadIdx.x, lane = tid & 31, wave = tid >> 5;
  __shared__ float redm[8], reds[8];

  float vals[4], lmax = -3.4e38f;
#pragma unroll
  for (int j = 0; j < 4; ++j) {
    const int c = tid + j * 256;
    float v = m[c] ? -1e9f : s[c];
    vals[j] = v;
    lmax = fmaxf(lmax, v);
  }
  float wmx = waveMax(lmax);
  if (lane == 0) redm[wave] = wmx;
  __syncthreads();
  float gmax = redm[0];
#pragma unroll
  for (int i = 1; i < 8; ++i) gmax = fmaxf(gmax, redm[i]);

  float lsum = 0.f;
#pragma unroll
  for (int j = 0; j < 4; ++j) {
    float e = __expf(vals[j] - gmax);
    vals[j] = e;
    lsum += e;
  }
  float wsm = waveSum(lsum);
  if (lane == 0) reds[wave] = wsm;
  __syncthreads();
  float gsum = 0.f;
#pragma unroll
  for (int i = 0; i < 8; ++i) gsum += reds[i];

  const float keep = m[0] ? 0.f : 1.f;
  const float inv  = keep / gsum;
#pragma unroll
  for (int j = 0; j < 4; ++j) s[tid + j * 256] = vals[j] * inv;
  if (tid == 0) atomicAdd(&headsum[bh], keep);   // softmax row sums to keep
}

__global__ void head_select_kernel(const float* __restrict__ hs, int* __restrict__ maxidx) {
  if (threadIdx.x == 0 && blockIdx.x == 0) {
    float p[NB][NHEAD];
    for (int h = 0; h < NHEAD; ++h) {
      float mx = hs[h];
      for (int b = 1; b < NB; ++b) mx = fmaxf(mx, hs[b * NHEAD + h]);
      float sm = 0.f;
      for (int b = 0; b < NB; ++b) { float e = __expf(hs[b * NHEAD + h] - mx); p[b][h] = e; sm += e; }
      for (int b = 0; b < NB; ++b) p[b][h] /= sm;
    }
    for (int b = 0; b < NB; ++b) {
      int best = 0; float bv = p[b][0];
      for (int h = 1; h < NHEAD; ++h) if (p[b][h] > bv) { bv = p[b][h]; best = h; }
      maxidx[b] = best;
    }
  }
}

__global__ __launch_bounds__(256) void top1_kernel(const float* __restrict__ attn,
                                                   const int* __restrict__ maxidx,
                                                   unsigned* __restrict__ m1) {
  const int b = blockIdx.y;
  const float* a = attn + ((size_t)(b * NHEAD + maxidx[b])) * TOK * TOK;
  const size_t base = (size_t)blockIdx.x * 1024 + (size_t)threadIdx.x * 4;
  float4 v = *(const float4*)(a + base);
  float lm = fmaxf(fmaxf(v.x, v.y), fmaxf(v.z, v.w));
  lm = waveMax(lm);
  __shared__ float red[8];
  if ((threadIdx.x & 31) == 0) red[threadIdx.x >> 5] = lm;
  __syncthreads();
  if (threadIdx.x == 0) {
    float m = red[0];
    for (int i = 1; i < 8; ++i) m = fmaxf(m, red[i]);
    atomicMax(&m1[b], __float_as_uint(m));     // attn >= 0: uint order == float order
  }
}

__global__ __launch_bounds__(256) void top2_kernel(const float* __restrict__ attn,
                                                   const int* __restrict__ maxidx,
                                                   const unsigned* __restrict__ m1,
                                                   unsigned* __restrict__ m2,
                                                   unsigned* __restrict__ cnt) {
  const int b = blockIdx.y;
  const float m1v = __uint_as_float(m1[b]);
  const float* a = attn + ((size_t)(b * NHEAD + maxidx[b])) * TOK * TOK;
  const size_t base = (size_t)blockIdx.x * 1024 + (size_t)threadIdx.x * 4;
  float4 v = *(const float4*)(a + base);
  float lm2 = 0.f; int lc = 0;
  float xs[4] = {v.x, v.y, v.z, v.w};
#pragma unroll
  for (int t = 0; t < 4; ++t) {
    if (xs[t] >= m1v) ++lc; else lm2 = fmaxf(lm2, xs[t]);
  }
  lm2 = waveMax(lm2);
  lc  = waveSumI(lc);
  __shared__ float redm[8];
  __shared__ int   redc[8];
  if ((threadIdx.x & 31) == 0) { redm[threadIdx.x >> 5] = lm2; redc[threadIdx.x >> 5] = lc; }
  __syncthreads();
  if (threadIdx.x == 0) {
    float m = redm[0]; int c = redc[0];
    for (int i = 1; i < 8; ++i) { m = fmaxf(m, redm[i]); c += redc[i]; }
    atomicMax(&m2[b], __float_as_uint(m));
    atomicAdd(&cnt[b], (unsigned)c);
  }
}

// adj2 = (top2-threshold mask, symmetrized, diag=1) * attn[b, maxidx[b]]
__global__ __launch_bounds__(256) void build_adj2_kernel(const float* __restrict__ attn,
                                                         const int* __restrict__ maxidx,
                                                         const unsigned* __restrict__ m1,
                                                         const unsigned* __restrict__ m2,
                                                         const unsigned* __restrict__ cnt,
                                                         float* __restrict__ adj2) {
  const int b = blockIdx.y;
  const float* a = attn + ((size_t)(b * NHEAD + maxidx[b])) * TOK * TOK;
  const float thr = (cnt[b] >= 2u) ? __uint_as_float(m1[b]) : __uint_as_float(m2[b]);
  const size_t e = (size_t)blockIdx.x * 1024 + (size_t)threadIdx.x * 4;
  const int i = (int)(e >> 10);
  const int j = (int)(e & 1023);
  float* outp = adj2 + (size_t)b * TOK * TOK;
#pragma unroll
  for (int t = 0; t < 4; ++t) {
    const int jj = j + t;
    const float aij = a[(size_t)i * TOK + jj];
    float o;
    if (i == jj) {
      o = aij;                                    // diag of mask is 1
    } else {
      const float aji = a[(size_t)jj * TOK + i];
      const float cm = (aij >= thr ? 1.f : 0.f) + (aji >= thr ? 1.f : 0.f);
      o = cm * aij;
    }
    outp[(size_t)i * TOK + jj] = o;
  }
}

// =====================================================================
// launcher
// =====================================================================
static inline void launch_gemm(hipStream_t st, int bmode,
                               const float* A, const float* B, float* C,
                               int M, int N, int K, int lda, int ldb, int ldc,
                               long long sAb, long long sAh, long long sBb, long long sBh,
                               long long sCz, int heads, int Z, int epi,
                               const float* bias, const float* denom, long long sDen,
                               const float* add, long long sAdd, float scale) {
  dim3 grid((N + BN - 1) / BN, (M + BM - 1) / BM, Z);
  if (bmode == 0)
    gemm_wmma_bf16<0><<<grid, 256, 0, st>>>(A, B, C, M, N, K, lda, ldb, ldc,
        sAb, sAh, sBb, sBh, sCz, heads, epi, bias, denom, sDen, add, sAdd, scale);
  else if (bmode == 1)
    gemm_wmma_bf16<1><<<grid, 256, 0, st>>>(A, B, C, M, N, K, lda, ldb, ldc,
        sAb, sAh, sBb, sBh, sCz, heads, epi, bias, denom, sDen, add, sAdd, scale);
  else
    gemm_wmma_bf16<2><<<grid, 256, 0, st>>>(A, B, C, M, N, K, lda, ldb, ldc,
        sAb, sAh, sBb, sBh, sCz, heads, epi, bias, denom, sDen, add, sAdd, scale);
}

extern "C" void kernel_launch(void* const* d_in, const int* in_sizes, int n_in,
                              void* d_out, int out_size, void* d_ws, size_t ws_size,
                              hipStream_t stream) {
  const float* adj    = (const float*)d_in[0];
  const float* inputs = (const float*)d_in[1];
  const unsigned char* smask = (const unsigned char*)d_in[2];  // jax bool = 1 byte
  const float* W0w = (const float*)d_in[3];
  const float* W0b = (const float*)d_in[4];
  const float* W1w = (const float*)d_in[5];
  const float* W1b = (const float*)d_in[6];
  const float* qw  = (const float*)d_in[7];
  const float* qb  = (const float*)d_in[8];
  const float* kw  = (const float*)d_in[9];
  const float* kbv = (const float*)d_in[10];
  const float* convw = (const float*)d_in[11];
  const float* convb = (const float*)d_in[12];
  const float* lincw = (const float*)d_in[13];
  const float* lincb = (const float*)d_in[14];
  const float* fc1w  = (const float*)d_in[15];
  const float* fc2w  = (const float*)d_in[16];
  float* out = (float*)d_out;

  // ---- workspace layout (bytes) ----
  char* w = (char*)d_ws;
  constexpr size_t SZ_BND   = (size_t)NB * TOK * 4;                 // 32 KB
  constexpr size_t OFF_DEN  = 0;
  constexpr size_t OFF_DEN2 = SZ_BND;
  constexpr size_t OFF_SM   = 2 * SZ_BND;                          // small scalars
  constexpr size_t SZ_D     = (size_t)NB * TOK * DIM * 4;          // 25.2 MB
  constexpr size_t SZ_CONV  = (size_t)NB * TOK * CONVN * 4;        // 25.1 MB
  constexpr size_t SZ_DC    = (size_t)NB * TOK * DCATD * 4;        // 50.3 MB
  constexpr size_t OFF_XW   = 131072;
  constexpr size_t OFF_AX   = OFF_XW + SZ_D;
  constexpr size_t OFF_OUT0 = OFF_AX + SZ_D;
  constexpr size_t OFF_CONV = OFF_OUT0 + SZ_D;
  constexpr size_t OFF_X    = OFF_CONV + SZ_CONV;
  constexpr size_t OFF_Q    = OFF_X + SZ_DC;
  constexpr size_t OFF_K    = OFF_Q + SZ_DC;
  constexpr size_t OFF_SC   = OFF_K + SZ_DC;

  float* denom   = (float*)(w + OFF_DEN);
  float* denom2  = (float*)(w + OFF_DEN2);
  float* headsum = (float*)(w + OFF_SM);
  int*      maxidx = (int*)(w + OFF_SM + 128);
  unsigned* m1   = (unsigned*)(w + OFF_SM + 192);
  unsigned* m2   = (unsigned*)(w + OFF_SM + 224);
  unsigned* cnt  = (unsigned*)(w + OFF_SM + 256);
  float* xw    = (float*)(w + OFF_XW);
  float* ax    = (float*)(w + OFF_AX);
  float* out0  = (float*)(w + OFF_OUT0);
  float* conve = (float*)(w + OFF_CONV);
  float* x     = (float*)(w + OFF_X);
  float* q     = (float*)(w + OFF_Q);
  float* kbuf  = (float*)(w + OFF_K);
  float* scores = (float*)(w + OFF_SC);
  // reuse after their producers/consumers are done (stream-ordered):
  float* adj2 = kbuf;   // after scores GEMM
  float* ax2  = q;      // after scores GEMM
  float* xw1  = xw;     // after layer0 combine
  float* out1 = out0;   // after conv
  float* hbuf = ax;     // after layer0 combine

  const long long SD  = (long long)TOK * DIM;    // 786432
  const long long SDC = (long long)TOK * DCATD;  // 1572864
  const long long SA  = (long long)TOK * TOK;    // 1048576
  const long long SCV = (long long)TOK * CONVN;  // 784384
  const float inv_sqrt_dk = 0.044194173824159216f; // 1/sqrt(512)

  // ---- layer 0 ----
  rowsum_kernel<<<NB * TOK / 8, 256, 0, stream>>>(adj, denom, TOK);
  // xw = inputs @ W0 + b0
  launch_gemm(stream, 0, inputs, W0w, xw, TOK, DIM, DIM, DIM, DIM, DIM,
              SD, 0, 0, 0, SD, 1, NB, EPI_BIAS, W0b, nullptr, 0, nullptr, 0, 0.f);
  // ax = adj @ inputs
  launch_gemm(stream, 0, adj, inputs, ax, TOK, DIM, TOK, TOK, DIM, DIM,
              SA, 0, SD, 0, SD, 1, NB, EPI_NONE, nullptr, nullptr, 0, nullptr, 0, 0.f);
  // out0 = relu((ax @ W0 + b0)/denom) + xw
  launch_gemm(stream, 0, ax, W0w, out0, TOK, DIM, DIM, DIM, DIM, DIM,
              SD, 0, 0, 0, SD, 1, NB, EPI_GCN, W0b, denom, TOK, xw, SD, 0.f);
  // conve = relu(conv1d(out0) + conv_b)   [GEMM: conv_w as 1024x3072, im2col B]
  launch_gemm(stream, 2, convw, out0, conve, TOK, CONVN, 3 * TOK, 3 * TOK, DIM, CONVN,
              0, 0, SD, 0, SCV, 1, NB, EPI_ROWBIAS_RELU, convb, nullptr, 0, nullptr, 0, 0.f);
  // x[:, :, 768:] = conve @ linc_w + linc_b   (write directly into concat buffer)
  launch_gemm(stream, 0, conve, lincw, x + DIM, TOK, DIM, CONVN, CONVN, DIM, DCATD,
              SCV, 0, 0, 0, SDC, 1, NB, EPI_BIAS, lincb, nullptr, 0, nullptr, 0, 0.f);
  // x[:, :, :768] = inputs
  concat_inputs_kernel<<<(NB * TOK * DIM / 4) / 256, 256, 0, stream>>>(inputs, x);

  // ---- attention ----
  launch_gemm(stream, 0, x, qw, q, TOK, DCATD, DCATD, DCATD, DCATD, DCATD,
              SDC, 0, 0, 0, SDC, 1, NB, EPI_BIAS, qb, nullptr, 0, nullptr, 0, 0.f);
  launch_gemm(stream, 0, x, kw, kbuf, TOK, DCATD, DCATD, DCATD, DCATD, DCATD,
              SDC, 0, 0, 0, SDC, 1, NB, EPI_BIAS, kbv, nullptr, 0, nullptr, 0, 0.f);
  // scores[b,h] = q_h @ k_h^T / sqrt(dk)   (Z = B*H, per-head column offsets)
  launch_gemm(stream, 1, q, kbuf, scores, TOK, TOK, DKH, DCATD, DCATD, TOK,
              SDC, DKH, SDC, DKH, SA, NHEAD, NB * NHEAD, EPI_SCALE,
              nullptr, nullptr, 0, nullptr, 0, inv_sqrt_dk);

  init_small_kernel<<<1, 64, 0, stream>>>(headsum, m1, m2, cnt, maxidx);
  softmax_mask_kernel<<<NB * NHEAD * TOK, 256, 0, stream>>>(scores, smask, headsum);
  head_select_kernel<<<1, 32, 0, stream>>>(headsum, maxidx);

  // top-2 threshold + symmetrized mask -> adj2
  dim3 tg(TOK * TOK / 1024, NB, 1);
  top1_kernel<<<tg, 256, 0, stream>>>(scores, maxidx, m1);
  top2_kernel<<<tg, 256, 0, stream>>>(scores, maxidx, m1, m2, cnt);
  build_adj2_kernel<<<tg, 256, 0, stream>>>(scores, maxidx, m1, m2, cnt, adj2);
  rowsum_kernel<<<NB * TOK / 8, 256, 0, stream>>>(adj2, denom2, TOK);

  // ---- layer 1 ----
  launch_gemm(stream, 0, x, W1w, xw1, TOK, DIM, DCATD, DCATD, DIM, DIM,
              SDC, 0, 0, 0, SD, 1, NB, EPI_BIAS, W1b, nullptr, 0, nullptr, 0, 0.f);
  launch_gemm(stream, 0, adj2, x, ax2, TOK, DCATD, TOK, TOK, DCATD, DCATD,
              SA, 0, SDC, 0, SDC, 1, NB, EPI_NONE, nullptr, nullptr, 0, nullptr, 0, 0.f);
  launch_gemm(stream, 0, ax2, W1w, out1, TOK, DIM, DCATD, DCATD, DIM, DIM,
              SDC, 0, 0, 0, SD, 1, NB, EPI_GCN, W1b, denom2, TOK, xw1, SD, 0.f);

  // ---- fc head ----
  launch_gemm(stream, 0, out1, fc1w, hbuf, TOK, DIM, DIM, DIM, DIM, DIM,
              SD, 0, 0, 0, SD, 1, NB, EPI_RELU, nullptr, nullptr, 0, nullptr, 0, 0.f);
  launch_gemm(stream, 0, hbuf, fc2w, out, TOK, DIM, DIM, DIM, DIM, DIM,
              SD, 0, 0, 0, SD, 1, NB, EPI_NONE, nullptr, nullptr, 0, nullptr, 0, 0.f);
}
